// BaseMLPExperts_24670292148244
// MI455X (gfx1250) — matrix operand order repaired and checked by reading, not compiled, not run on hardware
//
#include <hip/hip_runtime.h>

// Problem constants (match reference): E=8, H=2048, I=4096, G=1, C=2048
#define NE 8
#define NH 2048
#define NI 4096
#define NC 2048

// Tiling
#define TILE_M 128
#define TILE_N 128
#define TILE_K 64
#define APITCH 144  // bytes per LDS A row: 64 bf16 = 128B data + 16B pad (16B aligned, conflict-free)
#define BPITCH 144  // bytes per LDS B row (stored transposed [N][K])

typedef __attribute__((ext_vector_type(16))) __bf16 v16bf;
typedef __attribute__((ext_vector_type(8)))  __bf16 v8bf;
typedef __attribute__((ext_vector_type(8)))  float  v8f;

// Packed fp32->bf16 convert on the hardware datapath: one VALU op per dword.
static __device__ __forceinline__ unsigned pack2(float lo, float hi) {
    unsigned r;
    asm("v_cvt_pk_bf16_f32 %0, %1, %2" : "=v"(r) : "v"(lo), "v"(hi));
    return r;
}

// Branch-free exact-erf GELU (A&S 7.1.26, |err| < 1.5e-7): v_rcp + v_exp only.
static __device__ __forceinline__ float gelu_erf(float x) {
    float ax = fabsf(x) * 0.70710678118654752f;      // |x|/sqrt(2)
    float t  = __builtin_amdgcn_rcpf(1.0f + 0.3275911f * ax);
    float p  = t * (0.254829592f +
               t * (-0.284496736f +
               t * (1.421413741f +
               t * (-1.453152027f +
               t *  1.061405429f))));
    float e  = __expf(-ax * ax);
    float er = 1.0f - p * e;                          // erf(|x|/sqrt(2))
    er = __builtin_copysignf(er, x);
    return 0.5f * x * (1.0f + er);
}

#define SHUF16(lo, hi) __builtin_shufflevector((lo), (hi), 0,1,2,3,4,5,6,7,8,9,10,11,12,13,14,15)

// ---------------------------------------------------------------------------
// GEMM1: hmid[e,c,i] = gelu_exact( X[e,c,:] . Wi[e,:,i] ), bf16 output to ws
// ---------------------------------------------------------------------------
__global__ __launch_bounds__(256) void gemm1_gelu_kernel(
    const float* __restrict__ X, const float* __restrict__ Wi,
    unsigned short* __restrict__ Hmid)
{
    __shared__ __align__(16) unsigned char smA[2][TILE_M * APITCH];
    __shared__ __align__(16) unsigned char smB[2][TILE_N * BPITCH];

    const int n0 = blockIdx.x * TILE_N;
    const int c0 = blockIdx.y * TILE_M;
    const int e  = blockIdx.z;

    const float* Ax = X  + ((size_t)e * NC + c0) * (size_t)NH;
    const float* Bw = Wi + (size_t)e * NH * NI + n0;
    unsigned short* Hp = Hmid + ((size_t)e * NC + c0) * (size_t)NI + n0;

    const int tid  = threadIdx.x;
    const int lane = tid & 31;
    const int wave = tid >> 5;
    const int wm = (wave & 3) * 32;
    const int wn = (wave >> 2) * 64;
    const int lr = lane & 15;
    const int hh = lane >> 4;

    v8f acc[2][4] = {};

    // staged registers for next tile
    float4 aR[8], bR0[4], bR1[4];

    auto loadA = [&](int k0) {
        #pragma unroll
        for (int it = 0; it < 8; ++it) {              // 2048 float4 items
            int idx = tid + it * 256;
            int row = idx >> 4;
            int kq  = (idx & 15) << 2;
            aR[it] = *(const float4*)(Ax + (size_t)row * NH + k0 + kq);
        }
    };
    auto putA = [&](int buf) {
        #pragma unroll
        for (int it = 0; it < 8; ++it) {
            int idx = tid + it * 256;
            int row = idx >> 4;
            int kq  = (idx & 15) << 2;
            *(uint2*)(&smA[buf][row * APITCH + kq * 2]) =
                make_uint2(pack2(aR[it].x, aR[it].y), pack2(aR[it].z, aR[it].w));
        }
    };
    auto loadB = [&](int k0) {
        #pragma unroll
        for (int it = 0; it < 4; ++it) {              // 1024 items: k-pair x col-quad
            int idx = tid + it * 256;
            int kp  = idx >> 5;                        // 0..31
            int n4  = (idx & 31) << 2;
            const float* bp = Bw + (size_t)(k0 + 2 * kp) * NI + n4;
            bR0[it] = *(const float4*)(bp);
            bR1[it] = *(const float4*)(bp + NI);
        }
    };
    auto putB = [&](int buf) {
        #pragma unroll
        for (int it = 0; it < 4; ++it) {
            int idx = tid + it * 256;
            int kp  = idx >> 5;
            int n4  = (idx & 31) << 2;
            *(unsigned*)(&smB[buf][(n4 + 0) * BPITCH + kp * 4]) = pack2(bR0[it].x, bR1[it].x);
            *(unsigned*)(&smB[buf][(n4 + 1) * BPITCH + kp * 4]) = pack2(bR0[it].y, bR1[it].y);
            *(unsigned*)(&smB[buf][(n4 + 2) * BPITCH + kp * 4]) = pack2(bR0[it].z, bR1[it].z);
            *(unsigned*)(&smB[buf][(n4 + 3) * BPITCH + kp * 4]) = pack2(bR0[it].w, bR1[it].w);
        }
    };

    loadA(0); loadB(0);
    putA(0);  putB(0);
    __syncthreads();

    const int NIT = NH / TILE_K;
    for (int i = 0; i < NIT; ++i) {
        const int cur  = i & 1;
        const int nxt  = cur ^ 1;
        const bool more = (i + 1) < NIT;
        if (more) { loadA((i + 1) * TILE_K); loadB((i + 1) * TILE_K); }

        #pragma unroll
        for (int kk = 0; kk < 2; ++kk) {
            v16bf af[2], bf[4];
            #pragma unroll
            for (int ms = 0; ms < 2; ++ms) {
                const v8bf* p = (const v8bf*)&smA[cur][(wm + ms * 16 + lr) * APITCH + kk * 64 + hh * 16];
                af[ms] = SHUF16(p[0], p[2]);
            }
            #pragma unroll
            for (int ns = 0; ns < 4; ++ns) {
                const v8bf* p = (const v8bf*)&smB[cur][(wn + ns * 16 + lr) * BPITCH + kk * 64 + hh * 16];
                bf[ns] = SHUF16(p[0], p[2]);
            }
            #pragma unroll
            for (int ms = 0; ms < 2; ++ms)
                #pragma unroll
                for (int ns = 0; ns < 4; ++ns)
                    acc[ms][ns] = __builtin_amdgcn_wmma_f32_16x16x32_bf16(
                        false, af[ms], false, bf[ns],
                        (short)0, acc[ms][ns], false, false);
        }

        if (more) { putA(nxt); putB(nxt); __syncthreads(); }
    }

    #pragma unroll
    for (int ms = 0; ms < 2; ++ms)
        #pragma unroll
        for (int ns = 0; ns < 4; ++ns)
            #pragma unroll
            for (int r = 0; r < 8; ++r) {
                int m = wm + ms * 16 + (hh << 3) + r;
                int n = wn + ns * 16 + lr;
                float g = gelu_erf(acc[ms][ns][r]);
                Hp[(size_t)m * NI + n] = (unsigned short)(pack2(g, g) & 0xFFFFu);
            }
}

// ---------------------------------------------------------------------------
// GEMM2: out[e,c,h] = Hmid[e,c,:] . Wo[e,:,h], fp32 output
// A tile staged via gfx1250 async global->LDS path (ASYNCcnt).
// ---------------------------------------------------------------------------
__global__ __launch_bounds__(256) void gemm2_kernel(
    const unsigned short* __restrict__ Hmid, const float* __restrict__ Wo,
    float* __restrict__ Out)
{
    __shared__ __align__(16) unsigned char smA[2][TILE_M * APITCH];
    __shared__ __align__(16) unsigned char smB[2][TILE_N * BPITCH];

    const int n0 = blockIdx.x * TILE_N;
    const int c0 = blockIdx.y * TILE_M;
    const int e  = blockIdx.z;

    const unsigned short* Ah = Hmid + ((size_t)e * NC + c0) * (size_t)NI;
    const float* Bw = Wo + (size_t)e * NI * NH + n0;
    float* Op = Out + ((size_t)e * NC + c0) * (size_t)NH + n0;

    const int tid  = threadIdx.x;
    const int lane = tid & 31;
    const int wave = tid >> 5;
    const int wm = (wave & 3) * 32;
    const int wn = (wave >> 2) * 64;
    const int lr = lane & 15;
    const int hh = lane >> 4;

    v8f acc[2][4] = {};
    float4 bR0[4], bR1[4];

    // A tile (bf16, K contiguous): async copy straight into LDS, no VGPR staging.
    auto asyncA = [&](int k0, int buf) {
        #pragma unroll
        for (int it = 0; it < 4; ++it) {              // 1024 uint4 items (8 bf16 each)
            int idx = tid + it * 256;
            int row = idx >> 3;
            int ko  = (idx & 7) << 3;
            unsigned ldsoff = (unsigned)(uintptr_t)(&smA[buf][row * APITCH + ko * 2]);
            unsigned long long ga =
                (unsigned long long)(uintptr_t)(Ah + (size_t)row * NI + k0 + ko);
            asm volatile("global_load_async_to_lds_b128 %0, %1, off"
                         :: "v"(ldsoff), "v"(ga) : "memory");
        }
    };
    auto loadB = [&](int k0) {
        #pragma unroll
        for (int it = 0; it < 4; ++it) {
            int idx = tid + it * 256;
            int kp  = idx >> 5;
            int n4  = (idx & 31) << 2;
            const float* bp = Bw + (size_t)(k0 + 2 * kp) * NH + n4;
            bR0[it] = *(const float4*)(bp);
            bR1[it] = *(const float4*)(bp + NH);
        }
    };
    auto putB = [&](int buf) {
        #pragma unroll
        for (int it = 0; it < 4; ++it) {
            int idx = tid + it * 256;
            int kp  = idx >> 5;
            int n4  = (idx & 31) << 2;
            *(unsigned*)(&smB[buf][(n4 + 0) * BPITCH + kp * 4]) = pack2(bR0[it].x, bR1[it].x);
            *(unsigned*)(&smB[buf][(n4 + 1) * BPITCH + kp * 4]) = pack2(bR0[it].y, bR1[it].y);
            *(unsigned*)(&smB[buf][(n4 + 2) * BPITCH + kp * 4]) = pack2(bR0[it].z, bR1[it].z);
            *(unsigned*)(&smB[buf][(n4 + 3) * BPITCH + kp * 4]) = pack2(bR0[it].w, bR1[it].w);
        }
    };

    asyncA(0, 0);
    loadB(0);
    putB(0);
    asm volatile("s_wait_asynccnt 0" ::: "memory");
    __syncthreads();

    const int NIT = NI / TILE_K;
    for (int i = 0; i < NIT; ++i) {
        const int cur  = i & 1;
        const int nxt  = cur ^ 1;
        const bool more = (i + 1) < NIT;
        if (more) { asyncA((i + 1) * TILE_K, nxt); loadB((i + 1) * TILE_K); }

        #pragma unroll
        for (int kk = 0; kk < 2; ++kk) {
            v16bf af[2], bf[4];
            #pragma unroll
            for (int ms = 0; ms < 2; ++ms) {
                const v8bf* p = (const v8bf*)&smA[cur][(wm + ms * 16 + lr) * APITCH + kk * 64 + hh * 16];
                af[ms] = SHUF16(p[0], p[2]);
            }
            #pragma unroll
            for (int ns = 0; ns < 4; ++ns) {
                const v8bf* p = (const v8bf*)&smB[cur][(wn + ns * 16 + lr) * BPITCH + kk * 64 + hh * 16];
                bf[ns] = SHUF16(p[0], p[2]);
            }
            #pragma unroll
            for (int ms = 0; ms < 2; ++ms)
                #pragma unroll
                for (int ns = 0; ns < 4; ++ns)
                    acc[ms][ns] = __builtin_amdgcn_wmma_f32_16x16x32_bf16(
                        false, af[ms], false, bf[ns],
                        (short)0, acc[ms][ns], false, false);
        }

        if (more) {
            putB(nxt);
            asm volatile("s_wait_asynccnt 0" ::: "memory");
            __syncthreads();
        }
    }

    #pragma unroll
    for (int ms = 0; ms < 2; ++ms)
        #pragma unroll
        for (int ns = 0; ns < 4; ++ns)
            #pragma unroll
            for (int r = 0; r < 8; ++r) {
                int m = wm + ms * 16 + (hh << 3) + r;
                int n = wn + ns * 16 + lr;
                Op[(size_t)m * NH + n] = acc[ms][ns][r];
            }
}

extern "C" void kernel_launch(void* const* d_in, const int* in_sizes, int n_in,
                              void* d_out, int out_size, void* d_ws, size_t ws_size,
                              hipStream_t stream) {
    const float* x  = (const float*)d_in[0];   // (1, 8, 2048, 2048) fp32
    const float* wi = (const float*)d_in[1];   // (8, 2048, 4096) fp32
    const float* wo = (const float*)d_in[2];   // (8, 4096, 2048) fp32
    float* out = (float*)d_out;                // (1, 8, 2048, 2048) fp32

    unsigned short* hmid = (unsigned short*)d_ws;   // bf16 hmid: E*C*I*2 = 128 MiB

    dim3 blk(256);
    dim3 g1(NI / TILE_N, NC / TILE_M, NE);     // (32, 16, 8)
    dim3 g2(NH / TILE_N, NC / TILE_M, NE);     // (16, 16, 8)

    gemm1_gelu_kernel<<<g1, blk, 0, stream>>>(x, wi, hmid);
    gemm2_kernel<<<g2, blk, 0, stream>>>(hmid, wo, out);
}
